// TTNetSharedWithoutFeatureMap_78013785964680
// MI455X (gfx1250) — matrix operand order repaired and checked by reading, not compiled, not run on hardware
//
#include <hip/hip_runtime.h>

typedef __attribute__((ext_vector_type(16))) _Float16 v16h;
typedef __attribute__((ext_vector_type(8)))  float    v8f;
typedef __attribute__((ext_vector_type(4)))  float    v4f;
typedef __attribute__((ext_vector_type(4)))  unsigned int v4u;
typedef __attribute__((ext_vector_type(8)))  int      v8i;
typedef __attribute__((ext_vector_type(4)))  int      v4i;

#define B_    1024
#define T_    256
#define N_    128
#define R_    128
#define OUT_  10
#define K_    (N_*R_)     // 16384
#define BM    32          // batch rows per workgroup
#define THREADS 512       // 16 waves
#define SROW  36          // padded LDS row stride (floats): 16B-aligned chunks, conflict-free

// ---------- prep kernel 1: Wt[n][kg] = (f16) W_shared[kg*R + n] ----------
__global__ __launch_bounds__(256) void prep_w(const float* __restrict__ W,
                                              _Float16* __restrict__ Wt) {
  int idx = blockIdx.x * 256 + threadIdx.x;   // 0 .. R_*K_-1 (2M)
  int n   = idx >> 14;                        // / K_
  int kg  = idx & (K_ - 1);
  Wt[idx] = (_Float16)W[(size_t)kg * R_ + n];
}

// ---------- prep kernel 2: s0[k] = sum_j W_first[0,j,k];  WL[i,o] = sum_j W_last[i,j,o]
__global__ __launch_bounds__(256) void prep_vec(const float* __restrict__ Wf,
                                                const float* __restrict__ Wl,
                                                float* __restrict__ s0,
                                                float* __restrict__ WL) {
  int tid = threadIdx.x;
  if (tid < R_) {
    float a = 0.f;
    for (int j = 0; j < N_; ++j) a += Wf[j * R_ + tid];
    s0[tid] = a;
  }
  for (int idx = tid; idx < R_ * OUT_; idx += 256) {
    int i = idx / OUT_, o = idx - i * OUT_;
    float a = 0.f;
    for (int j = 0; j < N_; ++j) a += Wl[(size_t)i * N_ * OUT_ + j * OUT_ + o];
    WL[idx] = a;
  }
}

// ---- TDM: DMA one step's x tile (BM rows x 128 f32, row stride T*N) into LDS ----
__device__ __forceinline__ void tdm_load_x(const float* gsrc, uint32_t lds_addr) {
  uint64_t ga = (uint64_t)(uintptr_t)gsrc;
  v4u g0;
  g0[0] = 1u;                                   // count=1, user mode
  g0[1] = lds_addr;                             // D#.lds_addr (bytes)
  g0[2] = (uint32_t)ga;                         // global_addr[31:0]
  g0[3] = (uint32_t)((ga >> 32) & 0x01FFFFFFu)  // global_addr[56:32]
        | (2u << 30);                           // type = 2 ("image")
  v8i g1;
  g1[0] = (int)(2u << 16);                      // wg_mask=0, data_size=2 (4 bytes)
  g1[1] = (int)0x80000000u;                     // tensor_dim0 = 32768 (bits 79:48, low16)
  g1[2] = (int)(32u << 16);                     // tensor_dim0 hi16=0 | tensor_dim1=32 (low16)
  g1[3] = (int)(128u << 16);                    // tensor_dim1 hi16=0 | tile_dim0 = 128
  g1[4] = 32;                                   // tile_dim1 = 32 | tile_dim2 = 0
  g1[5] = 32768;                                // tensor_dim0_stride[31:0] = T_*N_
  g1[6] = 0;                                    // dim0_stride hi | dim1_stride lo
  g1[7] = 0;
  v4i z = {0, 0, 0, 0};                         // 2-D tile: groups 2/3 unused
#if defined(__clang_major__) && (__clang_major__ >= 23)
  v8i z8 = {0, 0, 0, 0, 0, 0, 0, 0};
  __builtin_amdgcn_tensor_load_to_lds(g0, g1, z, z, z8, 0);
#else
  __builtin_amdgcn_tensor_load_to_lds(g0, g1, z, z, 0);
#endif
}

// ---------- main scan kernel ----------
// Grid: B_/BM workgroups. Each WG: state slab sT[128 i][32 row] fp32 in LDS,
// 16 waves = 2 M-tiles x 8 N-tiles of the 32x128 new-state tile.
// Per step t:  master[b,k] = sum_i s[b,i] * ( sum_j x_t[b,j] * W[i,j,k] )
// inner GEMM in f16 WMMA (f32 accum), outer scale/accumulate in fp32 VALU.
// x_{t+1} is DMA'd by the Tensor Data Mover into a double-buffered LDS tile
// while step t computes; s_wait_tensorcnt piggybacks on the state barriers.
__global__ __launch_bounds__(THREADS) void tt_scan(const float* __restrict__ tensor,
                                                   const _Float16* __restrict__ Wt,
                                                   const float* __restrict__ s0,
                                                   const float* __restrict__ WL,
                                                   float* __restrict__ out) {
  __shared__ float sT[R_][SROW];                // [i][batch-row], padded
  __shared__ float xbuf[2][BM][N_];             // double-buffered x tile (TDM dest)

  const int tid   = threadIdx.x;
  const int wave  = tid >> 5;
  const int lane  = tid & 31;
  const int hi    = lane >> 4;                  // lane half (0/1)
  const int lo    = lane & 15;
  const int Mt    = wave >> 3;                  // 0..1  -> 16-row M-tile
  const int nt    = wave & 7;                   // 0..7  -> 16-col N-tile
  const int mbase = Mt * 16;
  const int nbase = nt * 16;
  const int b0    = blockIdx.x * BM;

  const float* xg0 = tensor + (size_t)b0 * (T_ * N_);   // x_t row block base (row stride T_*N_)

  // kick off DMA of x_0 while we initialize the state
  if (wave == 0)
    tdm_load_x(xg0, (uint32_t)(uintptr_t)&xbuf[0][0][0]);

  for (int idx = tid; idx < R_ * BM; idx += THREADS) {
    int i = idx >> 5, r = idx & (BM - 1);
    sT[i][r] = s0[i];
  }
  if (wave == 0) __builtin_amdgcn_s_wait_tensorcnt(0);
  __syncthreads();

  // A-fragment addressing (contiguous k order, not AMD's table order; B uses the same):
  //   lane<16: M=lane,    K = {0..15}
  //   lane>=16: M=lane-16, K = {16..31}
  const int arow = mbase + lo;                  // local batch row
  const int kp0  = hi * 16;

  // B-fragment base (32x16 f16 B layout): col n = nbase+lo,
  // K run = i*128 + jc*32 + hi*16 + [0..15]  (contiguous in Wt)
  const _Float16* bbase = Wt + (size_t)(nbase + lo) * K_ + hi * 16;

  const float* srow = &sT[0][mbase + 8 * hi];   // advance by SROW per i
  float*       drow = &sT[nbase + lo][mbase + 8 * hi];

  #pragma unroll 1
  for (int t = 0; t < T_; ++t) {
    // DMA next step's x tile into the other buffer (overlaps the i-loop)
    if (wave == 0 && t + 1 < T_)
      tdm_load_x(xg0 + (size_t)(t + 1) * N_,
                 (uint32_t)(uintptr_t)&xbuf[(t + 1) & 1][0][0]);

    // ---- build the four A-fragments from the LDS x tile ----
    const float* xb = &xbuf[t & 1][arow][kp0];
    v16h xa[4];
    #pragma unroll
    for (int jc = 0; jc < 4; ++jc) {
      const v4f p0 = *(const v4f*)(xb + jc * 32);
      const v4f p1 = *(const v4f*)(xb + jc * 32 + 4);
      const v4f p2 = *(const v4f*)(xb + jc * 32 + 8);
      const v4f p3 = *(const v4f*)(xb + jc * 32 + 12);
      #pragma unroll
      for (int e = 0; e < 4; ++e) {
        xa[jc][e]      = (_Float16)p0[e];
        xa[jc][4 + e]  = (_Float16)p1[e];
        xa[jc][8 + e]  = (_Float16)p2[e];
        xa[jc][12 + e] = (_Float16)p3[e];
      }
    }

    v8f master = {};                            // fp32 new-state tile
    const _Float16* bp = bbase;
    const float*    sp = srow;

    #pragma unroll 4
    for (int i = 0; i < R_; ++i) {
      v8f tmp = {};
      tmp = __builtin_amdgcn_wmma_f32_16x16x32_f16(false, xa[0], false,
              *(const v16h*)(bp +  0), (short)0, tmp, false, false);
      tmp = __builtin_amdgcn_wmma_f32_16x16x32_f16(false, xa[1], false,
              *(const v16h*)(bp + 32), (short)0, tmp, false, false);
      tmp = __builtin_amdgcn_wmma_f32_16x16x32_f16(false, xa[2], false,
              *(const v16h*)(bp + 64), (short)0, tmp, false, false);
      tmp = __builtin_amdgcn_wmma_f32_16x16x32_f16(false, xa[3], false,
              *(const v16h*)(bp + 96), (short)0, tmp, false, false);
      // fp32 per-row state scale: C layout row = v + 8*hi (+mbase)
      const v4f sa = *(const v4f*)(sp);
      const v4f sb = *(const v4f*)(sp + 4);
      #pragma unroll
      for (int v = 0; v < 4; ++v) master[v]     = fmaf(sa[v], tmp[v],     master[v]);
      #pragma unroll
      for (int v = 0; v < 4; ++v) master[4 + v] = fmaf(sb[v], tmp[4 + v], master[4 + v]);
      bp += N_;          // next i block of K
      sp += SROW;        // next i row of state
    }

    __syncthreads();     // all reads of old state done
    {
      v4f* d4 = (v4f*)drow;
      v4f w0, w1;
      #pragma unroll
      for (int v = 0; v < 4; ++v) { w0[v] = master[v]; w1[v] = master[4 + v]; }
      d4[0] = w0;        // ds_store_b128 x2, transposed into sT
      d4[1] = w1;
    }
    // x_{t+1} DMA must be complete before the next step consumes it
    if (wave == 0) __builtin_amdgcn_s_wait_tensorcnt(0);
    __syncthreads();     // new state + next x tile visible
  }

  // ---- tiny epilogue: out[b,o] = sum_i sT[i][r] * WL[i,o] ----
  if (tid < BM * OUT_) {
    int r = tid / OUT_, o = tid - r * OUT_;
    float a = 0.f;
    for (int i = 0; i < R_; ++i) a = fmaf(sT[i][r], WL[i * OUT_ + o], a);
    out[(size_t)(b0 + r) * OUT_ + o] = a;
  }
}

extern "C" void kernel_launch(void* const* d_in, const int* in_sizes, int n_in,
                              void* d_out, int out_size, void* d_ws, size_t ws_size,
                              hipStream_t stream) {
  const float* tensor  = (const float*)d_in[0];   // [B,T,N] f32
  const float* W_first = (const float*)d_in[1];   // [1,N,R] f32
  const float* W_shar  = (const float*)d_in[2];   // [R,N,R] f32
  const float* W_last  = (const float*)d_in[3];   // [R,N,OUT] f32
  float* out = (float*)d_out;

  _Float16* Wt = (_Float16*)d_ws;                              // 4,194,304 B
  float*    s0 = (float*)((char*)d_ws + (size_t)R_ * K_ * sizeof(_Float16));
  float*    WL = s0 + R_;

  prep_w  <<<(R_ * K_) / 256, 256, 0, stream>>>(W_shar, Wt);
  prep_vec<<<1, 256, 0, stream>>>(W_first, W_last, s0, WL);
  tt_scan <<<B_ / BM, THREADS, 0, stream>>>(tensor, Wt, s0, WL, out);
}
